// SelfAttention_36971078484098
// MI455X (gfx1250) — matrix operand order repaired and checked by reading, not compiled, not run on hardware
//
#include <hip/hip_runtime.h>
#include <cmath>

typedef __attribute__((ext_vector_type(16))) _Float16 v16h;
typedef __attribute__((ext_vector_type(8)))  float    v8f;

#define B_  4
#define C_  256
#define N_  2304   // 48*48
#define NH_ 8
#define HD_ 32
#define NT_ (N_/16)   // 144 position tiles
#define OT_ (C_/16)   // 16 channel tiles
#define IB_ (N_/32)   // 72 query blocks of 32 rows

union AFrag { v16h v; unsigned u[8]; };

// ---- WMMA fragment gathers (layouts per CDNA5 ISA 7.12.2) ------------------
// A-matrix 16x32 f16: lane L -> row = L%16, kh = (L/16)*8;
//   VGPR v in 0..3 : K = kh + 2v ; VGPR v in 4..7 : K = 16 + kh + 2(v-4)
__device__ __forceinline__ v16h frag_a_load(const _Float16* p, int ld) {
  int lane = threadIdx.x & 31;
  int r  = lane & 15;
  int kh = (lane >> 4) << 3;
  const unsigned* pr = reinterpret_cast<const unsigned*>(p + (size_t)r * ld);
  AFrag f;
#pragma unroll
  for (int i = 0; i < 4; ++i) f.u[i]     = pr[(kh >> 1) + i];
#pragma unroll
  for (int i = 0; i < 4; ++i) f.u[4 + i] = pr[((16 + kh) >> 1) + i];
  return f.v;
}

// B-matrix 32x16 f16: lane L -> col n = L%16, kb = (L/16)*8; VGPR v in 0..3: K = kb+2v; v in 4..7: K = 16+kb+2(v-4).
// p points at element [n0][k0] of a K-contiguous (i.e. B^T row-major) buffer.
__device__ __forceinline__ v16h frag_b_load(const _Float16* p, int ld) {
  int lane = threadIdx.x & 31;
  int n  = lane & 15;
  int kb = (lane >> 4) << 3;
  const unsigned* pr = reinterpret_cast<const unsigned*>(p + (size_t)n * ld + kb);
  AFrag f;
#pragma unroll
  for (int i = 0; i < 8; ++i) f.u[i] = pr[(i & 3) + ((i >> 2) << 3)];
  return f.v;
}

__device__ __forceinline__ v8f wmma_f16(v16h a, v16h b, v8f c) {
  return __builtin_amdgcn_wmma_f32_16x16x32_f16(false, a, false, b,
                                                (short)0, c, false, false);
}

// xor-shuffle reductions over each 16-lane half (rows of a C/D fragment)
__device__ __forceinline__ float half_max(float x) {
#pragma unroll
  for (int m = 1; m < 16; m <<= 1) x = fmaxf(x, __shfl_xor(x, m, 32));
  return x;
}
__device__ __forceinline__ float half_sum(float x) {
#pragma unroll
  for (int m = 1; m < 16; m <<= 1) x += __shfl_xor(x, m, 32);
  return x;
}

// ---- 1) x[b][c][n] f32 -> xT[b][n][c] f16 ----------------------------------
__global__ void cvt_x_kernel(const float* __restrict__ x, _Float16* __restrict__ xT) {
  int t = blockIdx.x * blockDim.x + threadIdx.x;
  if (t >= B_ * C_ * N_) return;
  int n = t % N_;
  int c = (t / N_) % C_;
  int b = t / (C_ * N_);
  xT[((size_t)b * N_ + n) * C_ + c] = (_Float16)x[t];
}

// ---- weights f32 -> f16 (Wq,Wk,Wv,Wo packed) -------------------------------
__global__ void cvt_w_kernel(const float* __restrict__ wq, const float* __restrict__ wk,
                             const float* __restrict__ wv, const float* __restrict__ wo,
                             _Float16* __restrict__ wh) {
  int t = blockIdx.x * blockDim.x + threadIdx.x;
  if (t >= 4 * C_ * C_) return;
  int which = t >> 16;           // C_*C_ == 65536
  int idx   = t & 65535;
  const float* src = which == 0 ? wq : which == 1 ? wk : which == 2 ? wv : wo;
  wh[t] = (_Float16)src[idx];
}

// ---- 2) Qt/Kt[b][n][o] = sum_c xT[b][n][c] * W[o][c] -----------------------
// Each wave: one 16-row stripe x 4 o-tiles (A fragment reused 4x per K step).
__global__ void gemm_xtw_kernel(const _Float16* __restrict__ xT,
                                const _Float16* __restrict__ wh,   // Wq at 0, Wk at C_*C_
                                _Float16* __restrict__ Qt,
                                _Float16* __restrict__ Kt) {
  int wid = blockIdx.x * (blockDim.x >> 5) + (threadIdx.x >> 5);
  const int tiles = B_ * NT_ * (OT_ / 4);           // 2304 per output
  if (wid >= 2 * tiles) return;
  int which = wid / tiles;
  int rem   = wid % tiles;
  int b  = rem / (NT_ * (OT_ / 4));
  int r2 = rem % (NT_ * (OT_ / 4));
  int nt = r2 / (OT_ / 4);
  int og = r2 % (OT_ / 4);                          // group of 4 o-tiles

  const _Float16* W   = wh + (size_t)which * C_ * C_;
  const _Float16* ap  = xT + ((size_t)b * N_ + nt * 16) * C_;
  _Float16*       out = (which == 0 ? Qt : Kt) + ((size_t)b * N_ + nt * 16) * C_;

  v8f acc[4] = {};
#pragma unroll
  for (int kc = 0; kc < C_ / 32; ++kc) {
    v16h af = frag_a_load(ap + kc * 32, C_);
#pragma unroll
    for (int t = 0; t < 4; ++t) {
      v16h bf = frag_b_load(W + (size_t)((og * 4 + t) * 16) * C_ + kc * 32, C_);
      acc[t] = wmma_f16(af, bf, acc[t]);
    }
  }
  int lane = threadIdx.x & 31;
  int n  = lane & 15;
  int m0 = (lane >> 4) << 3;
#pragma unroll
  for (int t = 0; t < 4; ++t)
#pragma unroll
    for (int v = 0; v < 8; ++v)
      out[(size_t)(m0 + v) * C_ + (og * 4 + t) * 16 + n] = (_Float16)acc[t][v];
}

// ---- 3) Vh[b][o][n] = sum_c Wv[o][c] * x[b][c][n] --------------------------
// Each wave: one 16-channel stripe x 4 position tiles (A = Wv reused 4x).
__global__ void gemm_wxt_kernel(const _Float16* __restrict__ xT,
                                const _Float16* __restrict__ wvh,
                                _Float16* __restrict__ Vh) {
  int wid = blockIdx.x * (blockDim.x >> 5) + (threadIdx.x >> 5);
  const int tiles = B_ * OT_ * (NT_ / 4);           // 2304
  if (wid >= tiles) return;
  int b  = wid / (OT_ * (NT_ / 4));
  int r2 = wid % (OT_ * (NT_ / 4));
  int ot = r2 / (NT_ / 4);
  int ng = r2 % (NT_ / 4);                          // group of 4 position tiles

  v8f acc[4] = {};
#pragma unroll
  for (int kc = 0; kc < C_ / 32; ++kc) {
    v16h af = frag_a_load(wvh + (size_t)(ot * 16) * C_ + kc * 32, C_);  // A = Wv rows o
#pragma unroll
    for (int t = 0; t < 4; ++t) {
      v16h bf = frag_b_load(xT + ((size_t)b * N_ + (ng * 4 + t) * 16) * C_ + kc * 32, C_);
      acc[t] = wmma_f16(af, bf, acc[t]);
    }
  }
  int lane = threadIdx.x & 31;
  int n  = lane & 15;           // position within tile
  int m0 = (lane >> 4) << 3;    // channel row within tile
#pragma unroll
  for (int t = 0; t < 4; ++t)
#pragma unroll
    for (int v = 0; v < 8; ++v)
      Vh[((size_t)b * C_ + ot * 16 + m0 + v) * N_ + (ng * 4 + t) * 16 + n] = (_Float16)acc[t][v];
}

// ---- online-softmax row update for one 16-row group ------------------------
__device__ __forceinline__ void softmax_update(v8f& s0, v8f& s1, v8f& a0, v8f& a1,
                                               float* mi, float* li,
                                               _Float16 (*ptrow)[32],
                                               int m0, int nl, float scale) {
#pragma unroll
  for (int v = 0; v < 8; ++v) {
    s0[v] *= scale; s1[v] *= scale;
    float mx   = half_max(fmaxf(s0[v], s1[v]));
    float mnew = fmaxf(mi[v], mx);
    float alpha = __expf(mi[v] - mnew);
    s0[v] = __expf(s0[v] - mnew);
    s1[v] = __expf(s1[v] - mnew);
    li[v] = li[v] * alpha + half_sum(s0[v] + s1[v]);
    a0[v] *= alpha;
    a1[v] *= alpha;
    mi[v] = mnew;
    ptrow[m0 + v][nl]      = (_Float16)s0[v];
    ptrow[m0 + v][16 + nl] = (_Float16)s1[v];
  }
}

// ---- 4) fused flash attention: one wave per (b, h, 32-row block) -----------
__global__ void attn_kernel(const _Float16* __restrict__ Qt,
                            const _Float16* __restrict__ Kt,
                            const _Float16* __restrict__ Vh,
                            _Float16* __restrict__ aT) {
  __shared__ _Float16 pt[4][32][32];                 // P staging, per wave
  int lw  = threadIdx.x >> 5;
  int wid = blockIdx.x * (blockDim.x >> 5) + lw;
  if (wid >= B_ * NH_ * IB_) return;
  int b  = wid / (NH_ * IB_);
  int r2 = wid % (NH_ * IB_);
  int h  = r2 / IB_;
  int ib = r2 % IB_;

  int lane = threadIdx.x & 31;
  int nl   = lane & 15;
  int m0   = (lane >> 4) << 3;
  const float scale = 0.17677669529663687f;          // 1/sqrt(32)

  // Two Q fragments (rows ib*32..+15, +16..+31), resident for the whole pass
  const _Float16* qp = Qt + ((size_t)(b * N_ + ib * 32) * C_) + h * HD_;
  v16h qf0 = frag_a_load(qp, C_);
  v16h qf1 = frag_a_load(qp + (size_t)16 * C_, C_);

  v8f acc00 = {}, acc01 = {}, acc10 = {}, acc11 = {};
  float mi[2][8], li[2][8];
#pragma unroll
  for (int g = 0; g < 2; ++g)
#pragma unroll
    for (int v = 0; v < 8; ++v) { mi[g][v] = -1e30f; li[g][v] = 0.f; }

  const _Float16* kbase = Kt + (size_t)b * N_ * C_ + h * HD_;
  const _Float16* vbase = Vh + ((size_t)b * C_ + h * HD_) * N_;

  for (int jb = 0; jb < N_ / 32; ++jb) {
    // S = Q (32x32) x K^T (32x32) as four 16x16 WMMAs
    v16h kb0 = frag_b_load(kbase + (size_t)(jb * 32) * C_, C_);
    v16h kb1 = frag_b_load(kbase + (size_t)(jb * 32 + 16) * C_, C_);
    if (jb + 1 < N_ / 32) {
      __builtin_prefetch(kbase + (size_t)((jb + 1) * 32) * C_, 0, 1);
      __builtin_prefetch(vbase + (jb + 1) * 32, 0, 1);
    }
    v8f z = {};
    v8f s00 = wmma_f16(qf0, kb0, z);
    v8f s01 = wmma_f16(qf0, kb1, z);
    v8f s10 = wmma_f16(qf1, kb0, z);
    v8f s11 = wmma_f16(qf1, kb1, z);

    softmax_update(s00, s01, acc00, acc01, mi[0], li[0], &pt[lw][0],  m0, nl, scale);
    softmax_update(s10, s11, acc10, acc11, mi[1], li[1], &pt[lw][16], m0, nl, scale);

    asm volatile("s_wait_dscnt 0" ::: "memory");     // LDS stores visible to regather
    v16h pf0 = frag_a_load(&pt[lw][0][0],  32);      // P rows 0..15, K = j(0..31)
    v16h pf1 = frag_a_load(&pt[lw][16][0], 32);      // P rows 16..31

    // acc += P (32x32) x V (32x32) -> four 16x16 WMMAs
    v16h vb0 = frag_b_load(vbase + jb * 32, N_);
    v16h vb1 = frag_b_load(vbase + (size_t)16 * N_ + jb * 32, N_);
    acc00 = wmma_f16(pf0, vb0, acc00);
    acc01 = wmma_f16(pf0, vb1, acc01);
    acc10 = wmma_f16(pf1, vb0, acc10);
    acc11 = wmma_f16(pf1, vb1, acc11);
  }

  // normalize and store attended^T[b][i][c] (c contiguous for the out-proj A)
  _Float16* ob = aT + (size_t)(b * N_ + ib * 32) * C_ + h * HD_;
#pragma unroll
  for (int v = 0; v < 8; ++v) {
    float i0 = 1.f / li[0][v];
    float i1 = 1.f / li[1][v];
    ob[(size_t)(m0 + v) * C_ + nl]           = (_Float16)(acc00[v] * i0);
    ob[(size_t)(m0 + v) * C_ + 16 + nl]      = (_Float16)(acc01[v] * i0);
    ob[(size_t)(16 + m0 + v) * C_ + nl]      = (_Float16)(acc10[v] * i1);
    ob[(size_t)(16 + m0 + v) * C_ + 16 + nl] = (_Float16)(acc11[v] * i1);
  }
}

// ---- 5) out = LN(Wo*att + x) per 16-position stripe ------------------------
__global__ void out_kernel(const _Float16* __restrict__ aT,
                           const _Float16* __restrict__ woh,
                           const float* __restrict__ x,
                           const float* __restrict__ gamma,
                           const float* __restrict__ beta,
                           float* __restrict__ out) {
  __shared__ float zbuf[16][C_];
  __shared__ float s_mu[16], s_rs[16];
  int b  = blockIdx.x / NT_;
  int nb = blockIdx.x % NT_;
  int w    = threadIdx.x >> 5;
  int lane = threadIdx.x & 31;
  int nl   = lane & 15;
  int m0   = (lane >> 4) << 3;

  // hoist the shared A stripe (16 positions x K=256) into registers once
  const _Float16* ap = aT + (size_t)(b * N_ + nb * 16) * C_;
  v16h af[C_ / 32];
#pragma unroll
  for (int kc = 0; kc < C_ / 32; ++kc) af[kc] = frag_a_load(ap + kc * 32, C_);

#pragma unroll
  for (int tt = 0; tt < 2; ++tt) {
    int ot = w + tt * 8;                              // this wave's o-tile
    v8f acc = {};
#pragma unroll
    for (int kc = 0; kc < C_ / 32; ++kc) {
      v16h bf = frag_b_load(woh + (size_t)(ot * 16) * C_ + kc * 32, C_);
      acc = wmma_f16(af[kc], bf, acc);
    }
#pragma unroll
    for (int v = 0; v < 8; ++v) {
      int o  = ot * 16 + nl;
      int nn = nb * 16 + m0 + v;
      zbuf[m0 + v][o] = acc[v] + x[((size_t)b * C_ + o) * N_ + nn];  // + residual
    }
  }
  __syncthreads();

  // per-position stats over 256 channels: wave w owns rows 2w (lanes 0-15), 2w+1
  {
    int r = 2 * w + (lane >> 4);
    float s = 0.f, sq = 0.f;
#pragma unroll
    for (int k = 0; k < C_ / 16; ++k) {
      float zv = zbuf[r][nl + 16 * k];
      s += zv; sq += zv * zv;
    }
    s  = half_sum(s);
    sq = half_sum(sq);
    float mu  = s * (1.f / C_);
    float var = sq * (1.f / C_) - mu * mu;
    if (nl == 0) { s_mu[r] = mu; s_rs[r] = rsqrtf(var + 1e-5f); }
  }
  __syncthreads();

  // write: thread t handles position r = t&15, channels c = (t>>4)+16k (coalesced on n)
  int r  = threadIdx.x & 15;
  int c0 = threadIdx.x >> 4;
  float mu = s_mu[r], rs = s_rs[r];
#pragma unroll
  for (int k = 0; k < C_ / 16; ++k) {
    int c = c0 + 16 * k;
    float zv = (zbuf[r][c] - mu) * rs * gamma[c] + beta[c];
    out[((size_t)b * C_ + c) * N_ + nb * 16 + r] = zv;
  }
}

// ---- host-side launch ------------------------------------------------------
extern "C" void kernel_launch(void* const* d_in, const int* in_sizes, int n_in,
                              void* d_out, int out_size, void* d_ws, size_t ws_size,
                              hipStream_t stream) {
  const float* x     = (const float*)d_in[0];
  const float* Wq    = (const float*)d_in[1];
  const float* Wk    = (const float*)d_in[2];
  const float* Wv    = (const float*)d_in[3];
  const float* Wo    = (const float*)d_in[4];
  const float* gamma = (const float*)d_in[5];
  const float* beta  = (const float*)d_in[6];
  float* out = (float*)d_out;

  const size_t bnc = (size_t)B_ * N_ * C_;          // 2,359,296 elements
  _Float16* ws = (_Float16*)d_ws;
  _Float16* xT = ws;                                // [B][N][C]
  _Float16* Wh = xT + bnc;                          // Wq,Wk,Wv,Wo f16 (4*C*C)
  _Float16* Qt = Wh + 4 * (size_t)C_ * C_;          // [B][N][C]
  _Float16* Kt = Qt + bnc;                          // [B][N][C]
  _Float16* Vh = Kt + bnc;                          // [B][C][N]
  _Float16* aT = Vh + bnc;                          // [B][N][C]

  cvt_x_kernel<<<(B_ * C_ * N_ + 255) / 256, 256, 0, stream>>>(x, xT);
  cvt_w_kernel<<<(4 * C_ * C_ + 255) / 256, 256, 0, stream>>>(Wq, Wk, Wv, Wo, Wh);

  {  // Q and K projections: 2 * 2304 wave-stripes, 4 waves per block
    int waves = 2 * B_ * NT_ * (OT_ / 4);
    gemm_xtw_kernel<<<(waves + 3) / 4, 128, 0, stream>>>(xT, Wh, Qt, Kt);
  }
  {  // V projection
    int waves = B_ * OT_ * (NT_ / 4);
    gemm_wxt_kernel<<<(waves + 3) / 4, 128, 0, stream>>>(xT, Wh + 2 * (size_t)C_ * C_, Vh);
  }
  {  // fused flash attention (32 query rows per wave)
    int waves = B_ * NH_ * IB_;
    attn_kernel<<<(waves + 3) / 4, 128, 0, stream>>>(Qt, Kt, Vh, aT);
  }
  // output projection + residual + LayerNorm
  out_kernel<<<B_ * NT_, 256, 0, stream>>>(aT, Wh + 3 * (size_t)C_ * C_, x, gamma, beta, out);
}
